// PAMA_24627342475935
// MI455X (gfx1250) — matrix operand order repaired and checked by reading, not compiled, not run on hardware
//
#include <hip/hip_runtime.h>
#include <hip/hip_fp16.h>

#define B_ 2
#define C_ 64
#define H_ 128
#define W_ 512
#define CG 16
#define MAXD 192
#define NBH (B_ * H_)

typedef __attribute__((ext_vector_type(16))) _Float16 v16h;
typedef __attribute__((ext_vector_type(8)))  float    v8f;

union FragH {
  v16h v;
  _Float16 h[16];
  unsigned u[8];
};

// A-matrix 16x32 f16 fragment: VGPR j, lane l -> row=l&15,
// k = 2*(j&3) + 16*(j>>2) + 8*(l>>4) + {0,1}.  kptr points at the
// contiguous K-run for this lane's row.
__device__ __forceinline__ void load_frag_a(FragH& f, const _Float16* kptr, int lane) {
  const int hi8 = (lane & 16) ? 8 : 0;
#pragma unroll
  for (int j = 0; j < 8; ++j) {
    const int k = ((j & 3) << 1) + ((j >> 2) << 4) + hi8;
    f.u[j] = *reinterpret_cast<const unsigned*>(kptr + k);
  }
}
// B-matrix 32x16 f16 fragment: VGPR j, lane l -> col=l&15,
// k = 2*(j&3) + 16*(j>>2) + 8*(l>>4) + {0,1}.
__device__ __forceinline__ void load_frag_b(FragH& f, const _Float16* kptr, int lane) {
  const int hi8 = (lane & 16) ? 8 : 0;
#pragma unroll
  for (int j = 0; j < 8; ++j) {
    const int k = ((j & 3) << 1) + ((j >> 2) << 4) + hi8;
    f.u[j] = *reinterpret_cast<const unsigned*>(kptr + k);
  }
}

// ---------------- stage 1: bn + first grouped 3x3 conv + relu ----------------
__global__ __launch_bounds__(256) void k_bn_conv1(
    const float* __restrict__ x, const float* __restrict__ gamma,
    const float* __restrict__ beta, const float* __restrict__ mean,
    const float* __restrict__ var, const float* __restrict__ w1,
    const float* __restrict__ b1, float* __restrict__ xh,
    float* __restrict__ r1)
{
  __shared__ float s_x[CG][18][20];
  __shared__ float s_w[CG][CG][9];
  __shared__ float s_scale[CG], s_shift[CG], s_b1[CG];
  const int g = blockIdx.z & 3, b = blockIdx.z >> 2;
  const int h0 = blockIdx.y * 16, w0 = blockIdx.x * 16;
  const int tx = threadIdx.x, ty = threadIdx.y;
  const int tid = ty * 16 + tx;
  if (tid < CG) {
    const int c = g * CG + tid;
    const float sc = gamma[c] * rsqrtf(var[c] + 1e-5f);
    s_scale[tid] = sc;
    s_shift[tid] = beta[c] - mean[c] * sc;
    s_b1[tid] = b1[c];
  }
  for (int i = tid; i < CG * CG * 9; i += 256)
    (&s_w[0][0][0])[i] = w1[g * CG * CG * 9 + i];
  __syncthreads();
  for (int i = tid; i < CG * 18 * 18; i += 256) {
    const int c = i / 324, rr = i % 324, hh = rr / 18, ww = rr % 18;
    const int gh = h0 + hh - 1, gw = w0 + ww - 1;
    float v = 0.f;
    if (gh >= 0 && gh < H_ && gw >= 0 && gw < W_)
      v = x[((size_t)(b * C_ + g * CG + c) * H_ + gh) * W_ + gw] * s_scale[c] + s_shift[c];
    s_x[c][hh][ww] = v;
  }
  __syncthreads();
  const int h = h0 + ty, w = w0 + tx;
#pragma unroll
  for (int c = 0; c < CG; ++c)
    xh[((size_t)(b * C_ + g * CG + c) * H_ + h) * W_ + w] = s_x[c][ty + 1][tx + 1];
  for (int co = 0; co < CG; ++co) {
    float acc = s_b1[co];
    for (int ci = 0; ci < CG; ++ci) {
      const float* wp = s_w[co][ci];
#pragma unroll
      for (int kh = 0; kh < 3; ++kh)
#pragma unroll
        for (int kw = 0; kw < 3; ++kw)
          acc = fmaf(s_x[ci][ty + kh][tx + kw], wp[kh * 3 + kw], acc);
    }
    r1[((size_t)(b * C_ + g * CG + co) * H_ + h) * W_ + w] = fmaxf(acc, 0.f);
  }
}

// ------- stage 2: second grouped 3x3 conv + residual + grouped 1x1 conv -------
__global__ __launch_bounds__(256) void k_conv2_pw(
    const float* __restrict__ xh, const float* __restrict__ r1,
    const float* __restrict__ w2, const float* __restrict__ b2,
    const float* __restrict__ pw, const float* __restrict__ pb,
    float* __restrict__ q)
{
  __shared__ float s_r[CG][18][20];
  __shared__ float s_w[CG][CG][9];
  __shared__ float s_pw[CG][CG];
  __shared__ float s_b2[CG], s_pb[CG];
  const int g = blockIdx.z & 3, b = blockIdx.z >> 2;
  const int h0 = blockIdx.y * 16, w0 = blockIdx.x * 16;
  const int tx = threadIdx.x, ty = threadIdx.y;
  const int tid = ty * 16 + tx;
  if (tid < CG) {
    s_b2[tid] = b2[g * CG + tid];
    s_pb[tid] = pb[g * CG + tid];
  }
  for (int i = tid; i < CG * CG * 9; i += 256)
    (&s_w[0][0][0])[i] = w2[g * CG * CG * 9 + i];
  for (int i = tid; i < CG * CG; i += 256)
    (&s_pw[0][0])[i] = pw[g * CG * CG + i];
  for (int i = tid; i < CG * 18 * 18; i += 256) {
    const int c = i / 324, rr = i % 324, hh = rr / 18, ww = rr % 18;
    const int gh = h0 + hh - 1, gw = w0 + ww - 1;
    float v = 0.f;
    if (gh >= 0 && gh < H_ && gw >= 0 && gw < W_)
      v = r1[((size_t)(b * C_ + g * CG + c) * H_ + gh) * W_ + gw];
    s_r[c][hh][ww] = v;
  }
  __syncthreads();
  const int h = h0 + ty, w = w0 + tx;
  float qa[CG];
#pragma unroll
  for (int o = 0; o < CG; ++o) qa[o] = s_pb[o];
  for (int co = 0; co < CG; ++co) {
    float acc = s_b2[co];
    for (int ci = 0; ci < CG; ++ci) {
      const float* wp = s_w[co][ci];
#pragma unroll
      for (int kh = 0; kh < 3; ++kh)
#pragma unroll
        for (int kw = 0; kw < 3; ++kw)
          acc = fmaf(s_r[ci][ty + kh][tx + kw], wp[kh * 3 + kw], acc);
    }
    const float t = xh[((size_t)(b * C_ + g * CG + co) * H_ + h) * W_ + w] + acc;
#pragma unroll
    for (int o = 0; o < CG; ++o) qa[o] = fmaf(s_pw[o][co], t, qa[o]);
  }
#pragma unroll
  for (int o = 0; o < CG; ++o)
    q[((size_t)(b * C_ + g * CG + o) * H_ + h) * W_ + w] = qa[o];
}

// ----- stage 3: subtract width-mean per (b,c,h), pack f16 as [b][h][w][c] -----
__global__ __launch_bounds__(256) void k_center_half(
    const float* __restrict__ q, _Float16* __restrict__ out)
{
  const int bh = blockIdx.x, b = bh / H_, h = bh % H_;
  const int tid = threadIdx.x;
  __shared__ float s_part[C_][4];
  __shared__ float s_mean[C_];
  const int c = tid & 63, qq = tid >> 6;
  const float* row = q + ((size_t)(b * C_ + c) * H_ + h) * W_;
  float s = 0.f;
  for (int w = qq * 128; w < qq * 128 + 128; ++w) s += row[w];
  s_part[c][qq] = s;
  __syncthreads();
  if (tid < C_)
    s_mean[tid] = (s_part[tid][0] + s_part[tid][1] + s_part[tid][2] + s_part[tid][3]) * (1.f / W_);
  __syncthreads();
  _Float16* o = out + (size_t)bh * W_ * C_;
  for (int i = tid; i < W_ * C_; i += 256) {
    const int w = i >> 6, cc = i & 63;
    o[i] = (_Float16)(q[((size_t)(b * C_ + cc) * H_ + h) * W_ + w] - s_mean[cc]);
  }
}

// ------------- pack raw x as f16 in [b][h][c][v] layout for warps -------------
__global__ __launch_bounds__(256) void k_x_half(
    const float* __restrict__ x, _Float16* __restrict__ out)
{
  const int idx = blockIdx.x;                 // (b*C + c)*H + h
  const int b = idx / (C_ * H_), r = idx % (C_ * H_), c = r / H_, h = r % H_;
  const float* src = x + (size_t)idx * W_;
  _Float16* dst = out + ((size_t)(b * H_ + h) * C_ + c) * W_;
  const int t = threadIdx.x;
  dst[t] = (_Float16)src[t];
  dst[t + 256] = (_Float16)src[t + 256];
}

// ---------------- score GEMM: S[w][v] = sum_c Q[w][c] K[v][c] ----------------
__global__ __launch_bounds__(256) void k_score(
    const _Float16* __restrict__ Qh, const _Float16* __restrict__ Kh,
    _Float16* __restrict__ S)
{
  const int bh = blockIdx.y;
  const int wave = threadIdx.x >> 5, lane = threadIdx.x & 31;
  const int tile = blockIdx.x * 8 + wave;       // 0..127
  const int m0 = (tile >> 2) * 16;              // w-tile
  const int nc = tile & 3;                      // 128-wide v chunk
  const _Float16* Qb = Qh + (size_t)bh * W_ * C_;
  const _Float16* Kb = Kh + (size_t)bh * W_ * C_;
  _Float16* Sb = S + (size_t)bh * W_ * W_;
  const int hi8 = (lane & 16) ? 8 : 0;
  FragH a0, a1;
  const _Float16* arow = Qb + (size_t)(m0 + (lane & 15)) * C_;
  load_frag_a(a0, arow, lane);
  load_frag_a(a1, arow + 32, lane);
  for (int nt = 0; nt < 8; ++nt) {
    const int n0 = nc * 128 + nt * 16;
    const _Float16* brow = Kb + (size_t)(n0 + (lane & 15)) * C_;
    FragH b0, b1;
    load_frag_b(b0, brow, lane);
    load_frag_b(b1, brow + 32, lane);
    v8f acc = {};
    acc = __builtin_amdgcn_wmma_f32_16x16x32_f16(false, a0.v, false, b0.v, (short)0, acc, false, false);
    acc = __builtin_amdgcn_wmma_f32_16x16x32_f16(false, a1.v, false, b1.v, (short)0, acc, false, false);
#pragma unroll
    for (int r = 0; r < 8; ++r)
      Sb[(size_t)(m0 + r + hi8) * W_ + n0 + (lane & 15)] = (_Float16)acc[r];
  }
}

// ------------------------- softmax stats (row over v) -------------------------
__global__ __launch_bounds__(256) void k_rowstats(
    const _Float16* __restrict__ S, float* __restrict__ rmax, float* __restrict__ rinv)
{
  const int row = blockIdx.x * 8 + (threadIdx.x >> 5);
  const int lane = threadIdx.x & 31;
  const _Float16* p = S + (size_t)row * W_;
  float vals[16];
#pragma unroll
  for (int i = 0; i < 16; ++i) vals[i] = (float)p[lane + 32 * i];
  float m = vals[0];
#pragma unroll
  for (int i = 1; i < 16; ++i) m = fmaxf(m, vals[i]);
  for (int off = 16; off; off >>= 1) m = fmaxf(m, __shfl_xor(m, off, 32));
  float s = 0.f;
#pragma unroll
  for (int i = 0; i < 16; ++i) s += __expf(vals[i] - m);
  for (int off = 16; off; off >>= 1) s += __shfl_xor(s, off, 32);
  if (lane == 0) { rmax[row] = m; rinv[row] = 1.f / s; }
}

// ----------------------- softmax stats (column over w) ------------------------
__global__ __launch_bounds__(256) void k_colstats(
    const _Float16* __restrict__ S, float* __restrict__ cmax, float* __restrict__ cinv)
{
  const int bh = blockIdx.x >> 1;
  const int v = ((blockIdx.x & 1) << 8) + threadIdx.x;
  const _Float16* p = S + (size_t)bh * W_ * W_ + v;
  float m = -3.0e38f;
  for (int w = 0; w < W_; ++w) {
    if ((w & 63) == 0 && w + 64 < W_) __builtin_prefetch(p + (size_t)(w + 64) * W_, 0, 0);
    m = fmaxf(m, (float)p[(size_t)w * W_]);
  }
  float s = 0.f;
  for (int w = 0; w < W_; ++w) s += __expf((float)p[(size_t)w * W_] - m);
  const int o = bh * W_ + v;
  cmax[o] = m;
  cinv[o] = 1.f / s;
}

// ---- warp GEMM (transposed): out[b,c,h,w] = sum_v softmax(S)[w,v] * X[c,v] ----
// colmode=0: M = row-softmax of S (M_r2l);  colmode=1: M = col-softmax (M_l2r).
__global__ __launch_bounds__(256) void k_warp(
    const _Float16* __restrict__ S, const _Float16* __restrict__ XH,
    const float* __restrict__ smax, const float* __restrict__ sinv,
    float* __restrict__ out, int colmode)
{
  const int bh = blockIdx.y, b = bh / H_, h = bh % H_;
  const int wave = threadIdx.x >> 5, lane = threadIdx.x & 31;
  const int ntile = blockIdx.x * 8 + wave;      // 0..31
  const int wcol = ntile * 16 + (lane & 15);
  const _Float16* Sb = S + (size_t)bh * W_ * W_;
  const _Float16* Xb = XH + (size_t)bh * C_ * W_;
  const float mx = smax[bh * W_ + wcol];
  const float iv = sinv[bh * W_ + wcol];
  const int hi16 = (lane & 16) ? 16 : 0;
  const int hi8 = (lane & 16) ? 8 : 0;
  v8f acc[4] = {};
  for (int k0 = 0; k0 < W_; k0 += 32) {
    FragH bf;
#pragma unroll
    for (int j = 0; j < 8; ++j) {
      const int k = k0 + ((j & 3) << 1) + ((j >> 2) << 4) + hi8;
      float s0, s1;
      if (colmode) {
        s0 = (float)Sb[(size_t)k * W_ + wcol];
        s1 = (float)Sb[(size_t)(k + 1) * W_ + wcol];
      } else {
        s0 = (float)Sb[(size_t)wcol * W_ + k];
        s1 = (float)Sb[(size_t)wcol * W_ + k + 1];
      }
      bf.h[2 * j]     = (_Float16)(__expf(s0 - mx) * iv);
      bf.h[2 * j + 1] = (_Float16)(__expf(s1 - mx) * iv);
    }
#pragma unroll
    for (int ct = 0; ct < 4; ++ct) {
      FragH af;
      load_frag_a(af, Xb + (size_t)(ct * 16 + (lane & 15)) * W_ + k0, lane);
      acc[ct] = __builtin_amdgcn_wmma_f32_16x16x32_f16(
          false, af.v, false, bf.v, (short)0, acc[ct], false, false);
    }
  }
#pragma unroll
  for (int ct = 0; ct < 4; ++ct) {
#pragma unroll
    for (int r = 0; r < 8; ++r) {
      const int c = ct * 16 + r + hi8;
      out[(((size_t)b * C_ + c) * H_ + h) * W_ + wcol] = acc[ct][r];
    }
  }
}

// ------------------ cost1 = M_r2l[:, :, :, :MAXD] + cost ----------------------
__global__ __launch_bounds__(256) void k_cost(
    const _Float16* __restrict__ S, const float* __restrict__ cost,
    const float* __restrict__ rmax, const float* __restrict__ rinv,
    float* __restrict__ out)
{
  const size_t i = (size_t)blockIdx.x * 256 + threadIdx.x;
  const int d = (int)(i % MAXD);
  const size_t row = i / MAXD;
  const float m = __expf((float)S[row * W_ + d] - rmax[row]) * rinv[row];
  out[i] = m + cost[i];
}

extern "C" void kernel_launch(void* const* d_in, const int* in_sizes, int n_in,
                              void* d_out, int out_size, void* d_ws, size_t ws_size,
                              hipStream_t stream)
{
  const float* x_left   = (const float*)d_in[0];
  const float* x_right  = (const float*)d_in[1];
  const float* cost     = (const float*)d_in[2];
  const float* bn_gamma = (const float*)d_in[3];
  const float* bn_beta  = (const float*)d_in[4];
  const float* bn_mean  = (const float*)d_in[5];
  const float* bn_var   = (const float*)d_in[6];
  const float* rb_w1    = (const float*)d_in[7];
  const float* rb_b1    = (const float*)d_in[8];
  const float* rb_w2    = (const float*)d_in[9];
  const float* rb_b2    = (const float*)d_in[10];
  const float* bq_w     = (const float*)d_in[11];
  const float* bq_b     = (const float*)d_in[12];
  const float* bs_w     = (const float*)d_in[13];
  const float* bs_b     = (const float*)d_in[14];

  char* ws = (char*)d_ws;
  float*    q_l  = (float*)(ws);                    // 32 MB
  float*    q_r  = (float*)(ws + 33554432ull);      // 32 MB
  float*    xh_l = (float*)(ws + 67108864ull);      // 32 MB (dies after conv2)
  float*    r1_l = (float*)(ws + 100663296ull);     // 32 MB (dies after conv2)
  float*    xh_r = (float*)(ws + 134217728ull);     // 32 MB
  float*    r1_r = (float*)(ws + 167772160ull);     // 32 MB
  _Float16* S    = (_Float16*)(ws + 67108864ull);   // 128 MB, reuses xh_l/r1_l/xh_r/r1_r
  _Float16* Qh   = (_Float16*)(ws + 201326592ull);  // 16 MB
  _Float16* Kh   = (_Float16*)(ws + 218103808ull);  // 16 MB
  _Float16* XlH  = (_Float16*)(ws + 234881024ull);  // 16 MB
  _Float16* XrH  = (_Float16*)(ws + 251658240ull);  // 16 MB
  float*    rmax = (float*)(ws + 268435456ull);
  float*    rinv = (float*)(ws + 268959744ull);
  float*    cmax = (float*)(ws + 269484032ull);
  float*    cinv = (float*)(ws + 270008320ull);

  float* out = (float*)d_out;
  float* out_lT = out;
  float* out_rT = out + 8388608;
  float* out_c  = out + 16777216;

  dim3 gConv(W_ / 16, H_ / 16, B_ * 4), bConv(16, 16, 1);
  k_bn_conv1<<<gConv, bConv, 0, stream>>>(x_left, bn_gamma, bn_beta, bn_mean, bn_var,
                                          rb_w1, rb_b1, xh_l, r1_l);
  k_bn_conv1<<<gConv, bConv, 0, stream>>>(x_right, bn_gamma, bn_beta, bn_mean, bn_var,
                                          rb_w1, rb_b1, xh_r, r1_r);
  k_conv2_pw<<<gConv, bConv, 0, stream>>>(xh_l, r1_l, rb_w2, rb_b2, bq_w, bq_b, q_l);
  k_conv2_pw<<<gConv, bConv, 0, stream>>>(xh_r, r1_r, rb_w2, rb_b2, bs_w, bs_b, q_r);
  k_center_half<<<NBH, 256, 0, stream>>>(q_l, Qh);
  k_center_half<<<NBH, 256, 0, stream>>>(q_r, Kh);
  k_x_half<<<B_ * C_ * H_, 256, 0, stream>>>(x_left, XlH);
  k_x_half<<<B_ * C_ * H_, 256, 0, stream>>>(x_right, XrH);
  k_score<<<dim3(16, NBH), 256, 0, stream>>>(Qh, Kh, S);
  k_rowstats<<<(NBH * W_) / 8, 256, 0, stream>>>(S, rmax, rinv);
  k_colstats<<<NBH * 2, 256, 0, stream>>>(S, cmax, cinv);
  k_warp<<<dim3(4, NBH), 256, 0, stream>>>(S, XrH, rmax, rinv, out_lT, 0);
  k_warp<<<dim3(4, NBH), 256, 0, stream>>>(S, XlH, cmax, cinv, out_rT, 1);
  k_cost<<<(NBH * W_ * MAXD) / 256, 256, 0, stream>>>(S, cost, rmax, rinv, out_c);

  (void)in_sizes; (void)n_in; (void)out_size; (void)ws_size;
}
